// GATNetHeadsChanged4LayersLeakyReLU_31628139168042
// MI455X (gfx1250) — matrix operand order repaired and checked by reading, not compiled, not run on hardware
//
#include <hip/hip_runtime.h>
#include <hip/hip_bf16.h>

// ---------------------------------------------------------------------------
// Types for CDNA5 WMMA
// ---------------------------------------------------------------------------
typedef __attribute__((ext_vector_type(16))) __bf16 v16bf;
typedef __attribute__((ext_vector_type(8)))  float  v8f;

union FragU { v16bf v; uint4 q[2]; unsigned x[8]; };

// f32 -> bf16 round-to-nearest-even (raw bits, avoids header API differences)
__device__ __forceinline__ unsigned short f2bf(float f) {
  unsigned int u = __float_as_uint(f);
  u += 0x7FFFu + ((u >> 16) & 1u);
  return (unsigned short)(u >> 16);
}
__device__ __forceinline__ float bf2f(unsigned short b) {
  return __uint_as_float(((unsigned int)b) << 16);
}
// Order-preserving float<->uint encoding for atomic float max
__device__ __forceinline__ unsigned fenc(float f) {
  unsigned u = __float_as_uint(f);
  return (u & 0x80000000u) ? ~u : (u | 0x80000000u);
}
__device__ __forceinline__ float fdec(unsigned u) {
  return (u & 0x80000000u) ? __uint_as_float(u & 0x7FFFFFFFu)
                           : __uint_as_float(~u);
}

// ---------------------------------------------------------------------------
// Generic elementwise helpers
// ---------------------------------------------------------------------------
__global__ void zero_u32_kernel(unsigned* __restrict__ p, long long n) {
  long long gid = (long long)blockIdx.x * blockDim.x + threadIdx.x;
  if (gid < n) p[gid] = 0u;
}

__global__ void f32_to_bf16_kernel(const float* __restrict__ x,
                                   unsigned short* __restrict__ y, long long n) {
  long long gid = (long long)blockIdx.x * blockDim.x + threadIdx.x;
  if (gid < n) y[gid] = f2bf(x[gid]);
}

// y = bf16( leaky_relu(x + bias[col], 0.01) )
__global__ void bias_lrelu_bf16_kernel(const float* __restrict__ x,
                                       const float* __restrict__ bias,
                                       unsigned short* __restrict__ y,
                                       long long n, int c) {
  long long gid = (long long)blockIdx.x * blockDim.x + threadIdx.x;
  if (gid >= n) return;
  float v = x[gid] + bias[gid % c];
  v = v > 0.f ? v : 0.01f * v;
  y[gid] = f2bf(v);
}

// ---------------------------------------------------------------------------
// bf16 WMMA GEMM:  C[M,Nc] = A[M,K] @ B[K,Nc] (+ bias[col])
// A,B bf16 row-major; C f32.  M%16==0, Nc%64==0, K%64==0.
// Block = 128 threads (4 waves). Block tile = 16 (M) x 64 (N).
//
// LDS tiles are stored K-pair-packed: one u32 = {lo: K=2p, hi: K=2p+1}.
// Per the CDNA5 16-bit WMMA VGPR layouts, each lane's A and B fragments are
// then two contiguous 16B runs -> ds_load_b128 x2 per operand.
// Double-buffered: stage(k+1) overlaps compute(k); one barrier per k-step.
// ---------------------------------------------------------------------------
__global__ __launch_bounds__(128) void gemm_bf16_wmma_kernel(
    const unsigned short* __restrict__ A, const unsigned short* __restrict__ B,
    const float* __restrict__ bias, float* __restrict__ C,
    int M, int Nc, int K) {
  // stride 20 dwords (80B): 16B aligned rows, conflict-free b128 frag reads
  __shared__ unsigned As4[2][16][20];  // [buf][m][kpair]
  __shared__ unsigned Bs4[2][64][20];  // [buf][n][kpair]

  const int tid  = threadIdx.x;
  const int wave = tid >> 5;
  const int lane = tid & 31;
  const int half = lane >> 4;   // K-half selector per 16-bit WMMA layout
  const int ml   = lane & 15;
  const int m0 = blockIdx.y * 16;
  const int n0 = blockIdx.x * 64;

  // staging indices
  const int ar  = tid >> 3;        // A row 0..15
  const int ac  = (tid & 7) * 4;   // A col (bf16) 0,4,..,28
  const int bpr = tid >> 3;        // B pair-row 0..15 (global rows 2bpr,2bpr+1)
  const int bc  = (tid & 7) * 8;   // B col group 0,8,..,56

  auto stage = [&](int buf, int k0) {
    // A: 4 contiguous bf16 (b64 global load) -> 2 packed pair dwords
    {
      const unsigned short* src = A + (size_t)(m0 + ar) * K + k0 + ac;
      unsigned long long v = *(const unsigned long long*)src;
      As4[buf][ar][(ac >> 1) + 0] = (unsigned)v;
      As4[buf][ar][(ac >> 1) + 1] = (unsigned)(v >> 32);
    }
    // B: two b128 row loads, packed into [n][kpair] dwords
    {
      const uint4 q0 = *(const uint4*)(B + (size_t)(k0 + 2 * bpr + 0) * Nc + n0 + bc);
      const uint4 q1 = *(const uint4*)(B + (size_t)(k0 + 2 * bpr + 1) * Nc + n0 + bc);
      const unsigned* a0 = (const unsigned*)&q0;
      const unsigned* a1 = (const unsigned*)&q1;
#pragma unroll
      for (int j = 0; j < 4; ++j) {
        Bs4[buf][bc + 2 * j + 0][bpr] = (a0[j] & 0xFFFFu) | (a1[j] << 16);
        Bs4[buf][bc + 2 * j + 1][bpr] = (a0[j] >> 16) | (a1[j] & 0xFFFF0000u);
      }
    }
  };

  v8f acc = {};
  stage(0, 0);
  __syncthreads();

  for (int k0 = 0; k0 < K; k0 += 32) {
    const int buf = (k0 >> 5) & 1;
    if (k0 + 32 < K) stage(buf ^ 1, k0 + 32);   // overlap next-tile staging
    if (k0 + 64 < K) {                          // prefetch two tiles ahead
      __builtin_prefetch(A + (size_t)(m0 + ar) * K + k0 + 64 + ac, 0, 0);
      __builtin_prefetch(B + (size_t)(k0 + 64 + 2 * bpr) * Nc + n0 + bc, 0, 0);
    }

    // A frag: lanes 0-15 hold M=ml, K in {0..7,16..23}; lanes 16-31 K+8.
    // pair columns: {4h..4h+3} then {8+4h..8+4h+3}  -> 2x ds_load_b128
    FragU af, bfr;
    af.q[0] = *(const uint4*)&As4[buf][ml][4 * half];
    af.q[1] = *(const uint4*)&As4[buf][ml][8 + 4 * half];
    // B frag: pair columns {4h..4h+3} then {8+4h..8+4h+3} at column nn -> 2x ds_load_b128
    const int nn = wave * 16 + ml;
    bfr.q[0] = *(const uint4*)&Bs4[buf][nn][4 * half];
    bfr.q[1] = *(const uint4*)&Bs4[buf][nn][8 + 4 * half];

    acc = __builtin_amdgcn_wmma_f32_16x16x32_bf16(
        false, af.v, false, bfr.v, (short)0, acc, false, false);
    __syncthreads();
  }

  // C/D layout: VGPR r -> M = r + 8*half, col = lane&15
  const int nc = n0 + wave * 16 + ml;
  const float bv = bias ? bias[nc] : 0.0f;
#pragma unroll
  for (int r = 0; r < 8; ++r) {
    int mm = m0 + r + 8 * half;
    C[(long long)mm * Nc + nc] = acc[r] + bv;
  }
}

// ---------------------------------------------------------------------------
// GAT attention coefficients: a_src[n,h] = sum_d h[n,h,d]*att_src[h,d]  (wave/node)
// ---------------------------------------------------------------------------
__global__ __launch_bounds__(256) void attn_coef_kernel(
    const float* __restrict__ h, const float* __restrict__ att_src,
    const float* __restrict__ att_dst, float* __restrict__ a_src,
    float* __restrict__ a_dst, int n) {
  int node = blockIdx.x * (blockDim.x >> 5) + (threadIdx.x >> 5);
  int lane = threadIdx.x & 31;
  if (node >= n) return;
  const float* hp = h + (long long)node * 512;
  float s0 = 0.f, s1 = 0.f, d0 = 0.f, d1 = 0.f;
  for (int d = lane; d < 512; d += 32) {
    float v  = hp[d];
    float as = att_src[d];
    float ad = att_dst[d];
    if (d < 256) { s0 += v * as; d0 += v * ad; }
    else         { s1 += v * as; d1 += v * ad; }
  }
#pragma unroll
  for (int m = 16; m >= 1; m >>= 1) {
    s0 += __shfl_xor(s0, m, 32);
    s1 += __shfl_xor(s1, m, 32);
    d0 += __shfl_xor(d0, m, 32);
    d1 += __shfl_xor(d1, m, 32);
  }
  if (lane == 0) {
    a_src[node * 2 + 0] = s0; a_src[node * 2 + 1] = s1;
    a_dst[node * 2 + 0] = d0; a_dst[node * 2 + 1] = d1;
  }
}

// ---------------------------------------------------------------------------
// Edge pass 1: e = leaky_relu(a_src[src]+a_dst[dst], 0.2); segment max (dst)
// ---------------------------------------------------------------------------
__global__ void edge_pass1_kernel(const int* __restrict__ ei, int E, int n,
                                  const float* __restrict__ a_src,
                                  const float* __restrict__ a_dst,
                                  float* __restrict__ ebuf,
                                  unsigned* __restrict__ emaxu) {
  int gid = blockIdx.x * blockDim.x + threadIdx.x;
  int total = (E + n) * 2;
  if (gid >= total) return;
  int i = gid >> 1, hh = gid & 1;
  int s, d;
  if (i < E) { s = ei[i]; d = ei[E + i]; } else { s = d = i - E; }
  float e = a_src[s * 2 + hh] + a_dst[d * 2 + hh];
  e = e > 0.f ? e : 0.2f * e;
  ebuf[gid] = e;
  atomicMax(emaxu + d * 2 + hh, fenc(e));
}

// Edge pass 2: ex = exp(e - emax[dst]); segment sum (dst)
__global__ void edge_pass2_kernel(const int* __restrict__ ei, int E, int n,
                                  float* __restrict__ ebuf,
                                  const unsigned* __restrict__ emaxu,
                                  float* __restrict__ denom) {
  int gid = blockIdx.x * blockDim.x + threadIdx.x;
  int total = (E + n) * 2;
  if (gid >= total) return;
  int i = gid >> 1, hh = gid & 1;
  int d;
  if (i < E) { d = ei[E + i]; } else { d = i - E; }
  float ex = __expf(ebuf[gid] - fdec(emaxu[d * 2 + hh]));
  ebuf[gid] = ex;
  atomicAdd(denom + d * 2 + hh, ex);
}

// Edge pass 3 (wave/edge): out[dst,:] += alpha * h[src,:]  (b128 row reads)
__global__ __launch_bounds__(256) void edge_aggregate_kernel(
    const int* __restrict__ ei, int E, int n, const float* __restrict__ ebuf,
    const float* __restrict__ denom, const float* __restrict__ h,
    float* __restrict__ out) {
  int edge = blockIdx.x * (blockDim.x >> 5) + (threadIdx.x >> 5);
  if (edge >= E + n) return;
  int lane = threadIdx.x & 31;
  int s, d;
  if (edge < E) { s = ei[edge]; d = ei[E + edge]; } else { s = d = edge - E; }
  float al0 = ebuf[edge * 2 + 0] / denom[d * 2 + 0];
  float al1 = ebuf[edge * 2 + 1] / denom[d * 2 + 1];
  const float4* hp4 = (const float4*)(h + (long long)s * 512);
  float* op = out + (long long)d * 512;
#pragma unroll
  for (int t = 0; t < 4; ++t) {
    int g = lane + 32 * t;            // float4 group 0..127
    float4 v = hp4[g];
    float a = (g < 64) ? al0 : al1;   // head boundary at c=256 (= group 64)
    int c = g * 4;
    atomicAdd(op + c + 0, a * v.x);
    atomicAdd(op + c + 1, a * v.y);
    atomicAdd(op + c + 2, a * v.z);
    atomicAdd(op + c + 3, a * v.w);
  }
}

// ---------------------------------------------------------------------------
// BatchNorm (training-mode batch stats): block per column
// ---------------------------------------------------------------------------
__global__ __launch_bounds__(256) void bn_stats_kernel(
    const float* __restrict__ x, int n, int c,
    float* __restrict__ mean, float* __restrict__ rstd) {
  __shared__ float ss[256], ss2[256];
  int col = blockIdx.x, tid = threadIdx.x;
  float s = 0.f, s2 = 0.f;
  for (int r = tid; r < n; r += 256) {
    float v = x[(long long)r * c + col];
    s += v; s2 += v * v;
  }
  ss[tid] = s; ss2[tid] = s2;
  __syncthreads();
  for (int st = 128; st > 0; st >>= 1) {
    if (tid < st) { ss[tid] += ss[tid + st]; ss2[tid] += ss2[tid + st]; }
    __syncthreads();
  }
  if (tid == 0) {
    float m = ss[0] / (float)n;
    float v = ss2[0] / (float)n - m * m;
    mean[col] = m;
    rstd[col] = rsqrtf(fmaxf(v, 0.f) + 1e-5f);
  }
}

// y_bf16 = leaky_relu(g*(x-mean)*rstd + beta, 0.01)
__global__ void bn_act_bf16_kernel(const float* __restrict__ x,
                                   const float* __restrict__ g,
                                   const float* __restrict__ beta,
                                   const float* __restrict__ mean,
                                   const float* __restrict__ rstd,
                                   unsigned short* __restrict__ y,
                                   long long n, int c) {
  long long gid = (long long)blockIdx.x * blockDim.x + threadIdx.x;
  if (gid >= n) return;
  int ci = (int)(gid % c);
  float v = g[ci] * (x[gid] - mean[ci]) * rstd[ci] + beta[ci];
  v = v > 0.f ? v : 0.01f * v;
  y[gid] = f2bf(v);
}

// ---------------------------------------------------------------------------
// dense3: y[n,3] = y3[n,64] @ w3[64,3] + b3; also sq[n] = |y|^2
// ---------------------------------------------------------------------------
__global__ void dense3_kernel(const unsigned short* __restrict__ y3,
                              const float* __restrict__ w3,
                              const float* __restrict__ b3,
                              float* __restrict__ y, float* __restrict__ sq,
                              int n) {
  int i = blockIdx.x * blockDim.x + threadIdx.x;
  if (i >= n) return;
  float o0 = b3[0], o1 = b3[1], o2 = b3[2];
  const unsigned short* yp = y3 + (long long)i * 64;
#pragma unroll 8
  for (int k = 0; k < 64; ++k) {
    float v = bf2f(yp[k]);
    o0 += v * w3[k * 3 + 0];
    o1 += v * w3[k * 3 + 1];
    o2 += v * w3[k * 3 + 2];
  }
  y[i * 3 + 0] = o0; y[i * 3 + 1] = o1; y[i * 3 + 2] = o2;
  sq[i] = o0 * o0 + o1 * o1 + o2 * o2;
}

// Pairwise Euclidean distances via gram trick (matches reference clamping)
__global__ __launch_bounds__(256) void pairdist_kernel(
    const float* __restrict__ y, const float* __restrict__ sq,
    float* __restrict__ out, int n) {
  int j = blockIdx.x * 32 + (threadIdx.x & 31);
  int i = blockIdx.y * 8 + (threadIdx.x >> 5);
  if (i >= n || j >= n) return;
  float xi = y[i * 3], yi = y[i * 3 + 1], zi = y[i * 3 + 2];
  float xj = y[j * 3], yj = y[j * 3 + 1], zj = y[j * 3 + 2];
  float dot = xi * xj + yi * yj + zi * zj;
  float d2 = sq[i] + sq[j] - 2.f * dot;
  d2 = d2 > 0.f ? d2 : 0.f;
  out[(long long)i * n + j] = (d2 > 1e-12f) ? sqrtf(d2) : 0.f;
}

// ---------------------------------------------------------------------------
// Host launch
// ---------------------------------------------------------------------------
static inline long long cdiv(long long a, long long b) { return (a + b - 1) / b; }

extern "C" void kernel_launch(void* const* d_in, const int* in_sizes, int n_in,
                              void* d_out, int out_size, void* d_ws, size_t ws_size,
                              hipStream_t stream) {
  const float* x        = (const float*)d_in[0];
  const int*   ei       = (const int*)d_in[1];
  const float* w_conv   = (const float*)d_in[2];
  const float* att_src  = (const float*)d_in[3];
  const float* att_dst  = (const float*)d_in[4];
  const float* b_conv   = (const float*)d_in[5];
  const float* w_a      = (const float*)d_in[6];
  const float* b_a      = (const float*)d_in[7];
  const float* g_a      = (const float*)d_in[8];
  const float* beta_a   = (const float*)d_in[9];
  const float* w_1      = (const float*)d_in[10];
  const float* b_1      = (const float*)d_in[11];
  const float* g_1      = (const float*)d_in[12];
  const float* beta_1   = (const float*)d_in[13];
  const float* w_2      = (const float*)d_in[14];
  const float* b_2      = (const float*)d_in[15];
  const float* g_2      = (const float*)d_in[16];
  const float* beta_2   = (const float*)d_in[17];
  const float* w_3      = (const float*)d_in[18];
  const float* b_3      = (const float*)d_in[19];

  const int N = in_sizes[0] / 512;      // 10000
  const int E = in_sizes[1] / 2;        // 320000

  // ---- workspace carve-out (aliased by lifetime) ----
  char* ws = (char*)d_ws;
  size_t off = 0;
  auto alloc = [&](size_t bytes) -> char* {
    char* p = ws + off;
    off = (off + bytes + 255) & ~(size_t)255;
    return p;
  };
  float*          h_buf  = (float*)alloc((size_t)N * 512 * 4);   // live: gemm1..e3
  float*          agg    = (float*)alloc((size_t)N * 512 * 4);   // live: zero..bias_lrelu
  unsigned short* xbf    = (unsigned short*)alloc((size_t)N * 512 * 2); // live: ..gemm1
  unsigned short* wbf    = (unsigned short*)alloc(512 * 512 * 2);
  unsigned short* wabf   = (unsigned short*)alloc(512 * 256 * 2);
  unsigned short* w1bf   = (unsigned short*)alloc(256 * 128 * 2);
  unsigned short* w2bf   = (unsigned short*)alloc(128 * 64 * 2);
  float*          a_src  = (float*)alloc((size_t)N * 2 * 4);
  float*          a_dst  = (float*)alloc((size_t)N * 2 * 4);
  unsigned*       emaxu  = (unsigned*)alloc((size_t)N * 2 * 4);
  float*          denom  = (float*)alloc((size_t)N * 2 * 4);
  float*          ebuf   = (float*)alloc((size_t)(E + N) * 2 * 4);
  float*          ybuf   = (float*)alloc((size_t)N * 3 * 4);
  float*          sqbuf  = (float*)alloc((size_t)N * 4);
  float*          bnp    = (float*)alloc(2 * (256 + 128 + 64) * 4);
  // aliases (dead-buffer reuse)
  float*          t1  = (float*)xbf;                                  // N*256 f32
  unsigned short* y0b = (unsigned short*)h_buf;                       // N*512 bf16
  float*          t2  = (float*)((char*)h_buf + (size_t)N * 512 * 2); // N*128 f32
  unsigned short* y1b = (unsigned short*)agg;                         // N*256 bf16
  unsigned short* y2b = (unsigned short*)((char*)agg + (size_t)N * 256 * 2); // N*128
  float*          t3  = (float*)((char*)agg + (size_t)N * 384 * 2);          // N*64 f32
  unsigned short* y3b = (unsigned short*)((char*)agg + (size_t)N * 512 * 2); // N*64
  float* mean_a = bnp,        * rstd_a = bnp + 256;
  float* mean_1 = bnp + 512,  * rstd_1 = bnp + 640;
  float* mean_2 = bnp + 768,  * rstd_2 = bnp + 832;

  float* dmat = (float*)d_out;

  // 1) bf16 conversions of GEMM operands
  {
    long long n;
    n = (long long)N * 512;
    f32_to_bf16_kernel<<<cdiv(n, 256), 256, 0, stream>>>(x, xbf, n);
    n = 512 * 512;
    f32_to_bf16_kernel<<<cdiv(n, 256), 256, 0, stream>>>(w_conv, wbf, n);
    n = 512 * 256;
    f32_to_bf16_kernel<<<cdiv(n, 256), 256, 0, stream>>>(w_a, wabf, n);
    n = 256 * 128;
    f32_to_bf16_kernel<<<cdiv(n, 256), 256, 0, stream>>>(w_1, w1bf, n);
    n = 128 * 64;
    f32_to_bf16_kernel<<<cdiv(n, 256), 256, 0, stream>>>(w_2, w2bf, n);
  }

  // 2) h = x @ w_conv   [N,512]
  gemm_bf16_wmma_kernel<<<dim3(512 / 64, N / 16), 128, 0, stream>>>(
      xbf, wbf, nullptr, h_buf, N, 512, 512);

  // 3) attention coefficients
  attn_coef_kernel<<<cdiv(N, 8), 256, 0, stream>>>(h_buf, att_src, att_dst,
                                                   a_src, a_dst, N);

  // 4) zero segment-max / denom / aggregation buffers
  zero_u32_kernel<<<cdiv((long long)N * 2, 256), 256, 0, stream>>>(emaxu, (long long)N * 2);
  zero_u32_kernel<<<cdiv((long long)N * 2, 256), 256, 0, stream>>>((unsigned*)denom, (long long)N * 2);
  zero_u32_kernel<<<cdiv((long long)N * 512, 256), 256, 0, stream>>>((unsigned*)agg, (long long)N * 512);

  // 5) edge softmax + aggregation
  {
    long long tot = (long long)(E + N) * 2;
    edge_pass1_kernel<<<cdiv(tot, 256), 256, 0, stream>>>(ei, E, N, a_src, a_dst, ebuf, emaxu);
    edge_pass2_kernel<<<cdiv(tot, 256), 256, 0, stream>>>(ei, E, N, ebuf, emaxu, denom);
    edge_aggregate_kernel<<<cdiv((long long)(E + N), 8), 256, 0, stream>>>(
        ei, E, N, ebuf, denom, h_buf, agg);
  }

  // 6) y0 = bf16(lrelu(agg + b_conv))
  bias_lrelu_bf16_kernel<<<cdiv((long long)N * 512, 256), 256, 0, stream>>>(
      agg, b_conv, y0b, (long long)N * 512, 512);

  // 7) densea: t1 = y0 @ w_a + b_a ; BN ; lrelu -> y1 (bf16)
  gemm_bf16_wmma_kernel<<<dim3(256 / 64, N / 16), 128, 0, stream>>>(
      y0b, wabf, b_a, t1, N, 256, 512);
  bn_stats_kernel<<<256, 256, 0, stream>>>(t1, N, 256, mean_a, rstd_a);
  bn_act_bf16_kernel<<<cdiv((long long)N * 256, 256), 256, 0, stream>>>(
      t1, g_a, beta_a, mean_a, rstd_a, y1b, (long long)N * 256, 256);

  // 8) dense1: t2 = y1 @ w_1 + b_1 ; BN ; lrelu -> y2 (bf16)
  gemm_bf16_wmma_kernel<<<dim3(128 / 64, N / 16), 128, 0, stream>>>(
      y1b, w1bf, b_1, t2, N, 128, 256);
  bn_stats_kernel<<<128, 256, 0, stream>>>(t2, N, 128, mean_1, rstd_1);
  bn_act_bf16_kernel<<<cdiv((long long)N * 128, 256), 256, 0, stream>>>(
      t2, g_1, beta_1, mean_1, rstd_1, y2b, (long long)N * 128, 128);

  // 9) dense2: t3 = y2 @ w_2 + b_2 ; BN ; lrelu -> y3 (bf16)
  gemm_bf16_wmma_kernel<<<dim3(64 / 64, N / 16), 128, 0, stream>>>(
      y2b, w2bf, b_2, t3, N, 64, 128);
  bn_stats_kernel<<<64, 256, 0, stream>>>(t3, N, 64, mean_2, rstd_2);
  bn_act_bf16_kernel<<<cdiv((long long)N * 64, 256), 256, 0, stream>>>(
      t3, g_2, beta_2, mean_2, rstd_2, y3b, (long long)N * 64, 64);

  // 10) dense3 + squared norms
  dense3_kernel<<<cdiv(N, 256), 256, 0, stream>>>(y3b, w_3, b_3, ybuf, sqbuf, N);

  // 11) pairwise distance matrix [N,N]
  pairdist_kernel<<<dim3(cdiv(N, 32), cdiv(N, 8)), 256, 0, stream>>>(
      ybuf, sqbuf, dmat, N);

  (void)n_in; (void)out_size; (void)ws_size;
}
